// EncoderProcesserDecoder_23416161698036
// MI455X (gfx1250) — matrix operand order repaired and checked by reading, not compiled, not run on hardware
//
#include <hip/hip_runtime.h>

#define NN 20000
#define NE 320000

typedef __attribute__((ext_vector_type(16))) __bf16 v16bf;
typedef __attribute__((ext_vector_type(8)))  float  v8f;

union Frag {
  v16bf v;
  unsigned int u[8];
  uint4 q[2];
};

struct MlpArgs {
  const __bf16* W1; const __bf16* W2; const __bf16* W3;   // padded bf16, row-major [Kpad x Npad]
  const float*  b1; const float*  b2; const float*  b3;
  const float*  g;  const float*  beta;
  const float*  in_f32;       // enc modes: raw features
  const __bf16* node;         // node latent (gather / input)
  __bf16*       edge;         // edge latent (in/out, residual in place)
  const float*  agg;          // node-mlp input aggregation
  float*        agg_out;      // edge-mlp atomic scatter target
  const int*    senders; const int* receivers;
  __bf16*       out_bf16;     // node latent store target
  float*        out_f32;      // decoder output
  int nRows; int k0raw;
};

// ---- one GEMM layer, 2 M-tiles per wave:
//   C[mt][16 x NT*16] = act(A[mt*16.. , KP] @ W[KP x NT*16] + b)
// B fragment loaded once per (kt,nt) and reused by both M-tiles.
template<int KP, int NT>
__device__ __forceinline__ void gemm_layer(const __bf16* __restrict__ sA,
                                           const __bf16* __restrict__ W,
                                           const float* __restrict__ bias,
                                           bool relu, v8f (&c)[2][NT]) {
  const int lane = threadIdx.x & 31;
  const int o2 = (lane >> 4) * 4;                 // upper half-wave: K+8 within each 16-group
  const unsigned int* arow0 = (const unsigned int*)(sA + (size_t)(lane & 15) * KP);
  const unsigned int* arow1 = (const unsigned int*)(sA + (size_t)(16 + (lane & 15)) * KP);
  #pragma unroll
  for (int mt = 0; mt < 2; ++mt)
    #pragma unroll
    for (int t = 0; t < NT; ++t)
      #pragma unroll
      for (int r = 0; r < 8; ++r) c[mt][t][r] = 0.f;

  for (int kt = 0; kt < KP / 32; ++kt) {
    Frag a0, a1;
    #pragma unroll
    for (int v = 0; v < 8; ++v) {                 // A frag: M=lane%16, K packed per ISA layout
      int idx = kt * 16 + o2 + (v & 3) + ((v >> 2) << 3);
      a0.u[v] = arow0[idx];
      a1.u[v] = arow1[idx];
    }
    const uint4* brow = (const uint4*)(W + (size_t)(kt * 32 + lane) * (NT * 16));
    #pragma unroll
    for (int t = 0; t < NT; ++t) {
      Frag b;                                     // B frag: N=lane%16, K per ISA layout (same as A)
      #pragma unroll
      for (int s = 0; s < 16; ++s)
        b.v[s] = W[(size_t)(kt * 32 + 2 * o2 + (s & 7) + ((s >> 3) << 4)) * (NT * 16) + t * 16 + (lane & 15)];
      c[0][t] = __builtin_amdgcn_wmma_f32_16x16x32_bf16(false, a0.v, false, b.v,
                                                        (short)0, c[0][t], false, false);
      c[1][t] = __builtin_amdgcn_wmma_f32_16x16x32_bf16(false, a1.v, false, b.v,
                                                        (short)0, c[1][t], false, false);
    }
  }
  #pragma unroll
  for (int t = 0; t < NT; ++t) {
    float bv = bias[t * 16 + (lane & 15)];
    #pragma unroll
    for (int mt = 0; mt < 2; ++mt)
      #pragma unroll
      for (int r = 0; r < 8; ++r) {
        float x = c[mt][t][r] + bv;
        c[mt][t][r] = relu ? fmaxf(x, 0.f) : x;
      }
  }
}

// store accumulators to LDS hidden buffer (bf16, pitch 128), per C/D layout
__device__ __forceinline__ void store_h(__bf16* __restrict__ sH, v8f (&c)[2][8]) {
  const int lane = threadIdx.x & 31;
  const int mb = (lane >> 4) * 8;
  #pragma unroll
  for (int mt = 0; mt < 2; ++mt)
    #pragma unroll
    for (int t = 0; t < 8; ++t) {
      int col = t * 16 + (lane & 15);
      #pragma unroll
      for (int r = 0; r < 8; ++r)
        sH[(size_t)(mt * 16 + mb + r) * 128 + col] = (__bf16)c[mt][t][r];
    }
}

// LayerNorm over 128 features, in registers: row M lives in 8 regs x 16 lanes (one half-wave)
__device__ __forceinline__ void layernorm(v8f (&c)[8], const float* __restrict__ g,
                                          const float* __restrict__ beta) {
  const int lane = threadIdx.x & 31;
  #pragma unroll
  for (int r = 0; r < 8; ++r) {
    float s1 = 0.f, s2 = 0.f;
    #pragma unroll
    for (int t = 0; t < 8; ++t) { float v = c[t][r]; s1 += v; s2 += v * v; }
    #pragma unroll
    for (int m = 1; m < 16; m <<= 1) {
      s1 += __shfl_xor(s1, m, 16);
      s2 += __shfl_xor(s2, m, 16);
    }
    float mu = s1 * (1.f / 128.f);
    float var = s2 * (1.f / 128.f) - mu * mu;
    float rs = rsqrtf(var + 1e-5f);
    #pragma unroll
    for (int t = 0; t < 8; ++t) {
      int col = t * 16 + (lane & 15);
      c[t][r] = (c[t][r] - mu) * rs * g[col] + beta[col];
    }
  }
}

// MODE: 0=enc_node 1=enc_edge 2=block_edge 3=block_node 4=decoder
// 128 threads = 4 waves; 32 rows per wave (2 M-tiles), 128 rows per workgroup.
template<int MODE, int KP0, int NT3, bool LN3>
__global__ __launch_bounds__(128) void mlp_kernel(MlpArgs A) {
  __shared__ __align__(16) __bf16 s_in[128 * KP0];
  __shared__ __align__(16) __bf16 s_h[128 * 128];
  const int wid  = threadIdx.x >> 5;
  const int lane = threadIdx.x & 31;
  const int rowbase = blockIdx.x * 128 + wid * 32;
  __bf16* sIn = s_in + (size_t)(wid * 32) * KP0;
  __bf16* sH  = s_h  + (size_t)(wid * 32) * 128;
  const int m  = lane & 15;
  const int hh = lane >> 4;
  const uint4 z4 = make_uint4(0u, 0u, 0u, 0u);

  // ---------------- stage input rows into LDS (bf16, K zero-padded) ----------------
  #pragma unroll
  for (int rr = 0; rr < 2; ++rr) {
    const int R = rowbase + rr * 16 + m;          // staging row for this lane
    const bool valid = R < A.nRows;
    __bf16* srow = sIn + (size_t)(rr * 16 + m) * KP0;
    if (MODE == 0 || MODE == 1) {
      for (int k = hh * 16; k < hh * 16 + 16; ++k) {
        float v = (valid && k < A.k0raw) ? A.in_f32[(size_t)R * A.k0raw + k] : 0.f;
        srow[k] = (__bf16)v;
      }
    } else if (MODE == 2) {                       // e_cat = [node[s] | node[r] | edge]
      int s = A.senders[R];
      int r = A.receivers[R];
      const uint4* ns = (const uint4*)(A.node + (size_t)s * 128);
      const uint4* nr = (const uint4*)(A.node + (size_t)r * 128);
      const uint4* ee = (const uint4*)(A.edge + (size_t)R * 128);
      uint4* drow = (uint4*)srow;
      for (int q = hh * 24; q < hh * 24 + 24; ++q) {
        uint4 val;
        if (q < 16) val = ns[q];
        else if (q < 32) val = nr[q - 16];
        else val = ee[q - 32];
        drow[q] = val;
      }
    } else if (MODE == 3) {                       // [node | agg]
      const uint4* nd = (const uint4*)(A.node + (size_t)R * 128);
      uint4* drow = (uint4*)srow;
      for (int q = hh * 8; q < hh * 8 + 8; ++q)
        drow[q] = valid ? nd[q] : z4;
      for (int k = hh * 64; k < hh * 64 + 64; ++k)
        srow[128 + k] = valid ? (__bf16)A.agg[(size_t)R * 128 + k] : (__bf16)0.f;
    } else {                                      // decoder: node latent
      const uint4* nd = (const uint4*)(A.node + (size_t)R * 128);
      uint4* drow = (uint4*)srow;
      for (int q = hh * 8; q < hh * 8 + 8; ++q)
        drow[q] = valid ? nd[q] : z4;
    }
  }
  __syncthreads();

  // ---------------- fused 3-layer MLP ----------------
  v8f c[2][8];
  gemm_layer<KP0, 8>(sIn, A.W1, A.b1, true, c);
  store_h(sH, c);
  __syncthreads();
  gemm_layer<128, 8>(sH, A.W2, A.b2, true, c);
  __syncthreads();
  store_h(sH, c);
  __syncthreads();
  v8f c3[2][NT3];
  gemm_layer<128, NT3>(sH, A.W3, A.b3, false, c3);
  if constexpr (LN3) {
    layernorm(c3[0], A.g, A.beta);
    layernorm(c3[1], A.g, A.beta);
  }

  // ---------------- epilogue ----------------
  int rv[2][8];
  if (MODE == 2) {
    #pragma unroll
    for (int mt = 0; mt < 2; ++mt)
      #pragma unroll
      for (int r = 0; r < 8; ++r)
        rv[mt][r] = A.receivers[rowbase + mt * 16 + hh * 8 + r];
  }
  #pragma unroll
  for (int mt = 0; mt < 2; ++mt) {
    #pragma unroll
    for (int t = 0; t < NT3; ++t) {
      int col = t * 16 + (lane & 15);
      #pragma unroll
      for (int r = 0; r < 8; ++r) {
        int RR = rowbase + mt * 16 + hh * 8 + r;  // C/D layout: M = r + 8*half
        float v = c3[mt][t][r];
        if (MODE == 0 || MODE == 1) {
          if (RR < A.nRows) A.out_bf16[(size_t)RR * 128 + col] = (__bf16)v;
        } else if (MODE == 2) {
          size_t p = (size_t)RR * 128 + col;
          float oldv = (float)A.edge[p];
          A.edge[p] = (__bf16)(oldv + v);         // edge residual in place
          atomicAdd(&A.agg_out[(size_t)rv[mt][r] * 128 + col], v);  // scatter-add e_new
        } else if (MODE == 3) {
          if (RR < A.nRows) {
            size_t p = (size_t)RR * 128 + col;
            float oldv = (float)A.out_bf16[p];
            A.out_bf16[p] = (__bf16)(oldv + v);   // node residual in place
          }
        } else {                                  // decoder: 2 output cols
          if (RR < A.nRows && (lane & 15) < 2)
            A.out_f32[(size_t)RR * 2 + col] = v;
        }
      }
    }
  }
}

// ---- weight prep: f32 [K x N] -> zero-padded bf16 row-major [KP x NP] ----
__global__ void conv_weight(const float* __restrict__ W, __bf16* __restrict__ dst,
                            int K, int N, int KP, int NP) {
  int idx = blockIdx.x * 256 + threadIdx.x;
  if (idx >= KP * NP) return;
  int r = idx / NP, cc = idx % NP;
  dst[idx] = (r < K && cc < N) ? (__bf16)W[(size_t)r * N + cc] : (__bf16)0.f;
}

__global__ void pad_bias(const float* __restrict__ b, float* __restrict__ dst, int n, int np) {
  int i = threadIdx.x;
  if (i < np) dst[i] = (i < n) ? b[i] : 0.f;
}

__global__ void zero_f32(float* p, int n) {
  int i = blockIdx.x * blockDim.x + threadIdx.x;
  if (i < n) p[i] = 0.f;
}

struct MlpPtrs { const float* W[3]; const float* b[3]; const float* g; const float* beta; };

extern "C" void kernel_launch(void* const* d_in, const int* in_sizes, int n_in,
                              void* d_out, int out_size, void* d_ws, size_t ws_size,
                              hipStream_t stream) {
  (void)in_sizes; (void)n_in; (void)out_size; (void)ws_size;

  // ---- unpack params (pytree insertion order: params, x, edge_attr, edge_index) ----
  int i = 0;
  auto take = [&](bool ln) {
    MlpPtrs p;
    for (int j = 0; j < 3; ++j) { p.W[j] = (const float*)d_in[i++]; p.b[j] = (const float*)d_in[i++]; }
    if (ln) { p.g = (const float*)d_in[i++]; p.beta = (const float*)d_in[i++]; }
    else    { p.g = nullptr; p.beta = nullptr; }
    return p;
  };
  MlpPtrs encN = take(true), encE = take(true);
  MlpPtrs blkE[4], blkN[4];
  for (int b = 0; b < 4; ++b) { blkE[b] = take(true); blkN[b] = take(true); }
  MlpPtrs dec = take(false);
  const float* x     = (const float*)d_in[i++];
  const float* eattr = (const float*)d_in[i++];
  const int*   eidx  = (const int*)d_in[i++];
  const int* senders   = eidx;
  const int* receivers = eidx + NE;

  // ---- workspace carving ----
  size_t off = 0;
  auto carve = [&](size_t bytes) {
    off = (off + 255) & ~(size_t)255;
    void* p = (char*)d_ws + off;
    off += bytes;
    return p;
  };
  auto convW = [&](const float* W, int K, int N, int KP, int NP) -> __bf16* {
    __bf16* dst = (__bf16*)carve((size_t)KP * NP * sizeof(__bf16));
    int tot = KP * NP;
    conv_weight<<<(tot + 255) / 256, 256, 0, stream>>>(W, dst, K, N, KP, NP);
    return dst;
  };

  __bf16 *encNw[3], *encEw[3], *blkEw[4][3], *blkNw[4][3], *decw[3];
  encNw[0] = convW(encN.W[0], 11, 128, 32, 128);
  encNw[1] = convW(encN.W[1], 128, 128, 128, 128);
  encNw[2] = convW(encN.W[2], 128, 128, 128, 128);
  encEw[0] = convW(encE.W[0], 3, 128, 32, 128);
  encEw[1] = convW(encE.W[1], 128, 128, 128, 128);
  encEw[2] = convW(encE.W[2], 128, 128, 128, 128);
  for (int b = 0; b < 4; ++b) {
    blkEw[b][0] = convW(blkE[b].W[0], 384, 128, 384, 128);
    blkEw[b][1] = convW(blkE[b].W[1], 128, 128, 128, 128);
    blkEw[b][2] = convW(blkE[b].W[2], 128, 128, 128, 128);
    blkNw[b][0] = convW(blkN[b].W[0], 256, 128, 256, 128);
    blkNw[b][1] = convW(blkN[b].W[1], 128, 128, 128, 128);
    blkNw[b][2] = convW(blkN[b].W[2], 128, 128, 128, 128);
  }
  decw[0] = convW(dec.W[0], 128, 128, 128, 128);
  decw[1] = convW(dec.W[1], 128, 128, 128, 128);
  decw[2] = convW(dec.W[2], 128, 2, 128, 16);
  float* decb3 = (float*)carve(16 * sizeof(float));
  pad_bias<<<1, 16, 0, stream>>>(dec.b[2], decb3, 2, 16);

  __bf16* nodeL = (__bf16*)carve((size_t)NN * 128 * sizeof(__bf16));
  __bf16* edgeL = (__bf16*)carve((size_t)NE * 128 * sizeof(__bf16));
  float*  agg   = (float*)carve((size_t)NN * 128 * sizeof(float));

  const int gN = (NN + 127) / 128, gE = NE / 128;

  // ---- encoders ----
  {
    MlpArgs a{};
    a.W1 = encNw[0]; a.W2 = encNw[1]; a.W3 = encNw[2];
    a.b1 = encN.b[0]; a.b2 = encN.b[1]; a.b3 = encN.b[2];
    a.g = encN.g; a.beta = encN.beta;
    a.in_f32 = x; a.out_bf16 = nodeL; a.nRows = NN; a.k0raw = 11;
    mlp_kernel<0, 32, 8, true><<<gN, 128, 0, stream>>>(a);
  }
  {
    MlpArgs a{};
    a.W1 = encEw[0]; a.W2 = encEw[1]; a.W3 = encEw[2];
    a.b1 = encE.b[0]; a.b2 = encE.b[1]; a.b3 = encE.b[2];
    a.g = encE.g; a.beta = encE.beta;
    a.in_f32 = eattr; a.out_bf16 = edgeL; a.nRows = NE; a.k0raw = 3;
    mlp_kernel<1, 32, 8, true><<<gE, 128, 0, stream>>>(a);
  }

  // ---- message passing blocks ----
  for (int b = 0; b < 4; ++b) {
    zero_f32<<<(NN * 128 + 255) / 256, 256, 0, stream>>>(agg, NN * 128);
    {
      MlpArgs a{};
      a.W1 = blkEw[b][0]; a.W2 = blkEw[b][1]; a.W3 = blkEw[b][2];
      a.b1 = blkE[b].b[0]; a.b2 = blkE[b].b[1]; a.b3 = blkE[b].b[2];
      a.g = blkE[b].g; a.beta = blkE[b].beta;
      a.node = nodeL; a.edge = edgeL; a.agg_out = agg;
      a.senders = senders; a.receivers = receivers;
      a.nRows = NE;
      mlp_kernel<2, 384, 8, true><<<gE, 128, 0, stream>>>(a);
    }
    {
      MlpArgs a{};
      a.W1 = blkNw[b][0]; a.W2 = blkNw[b][1]; a.W3 = blkNw[b][2];
      a.b1 = blkN[b].b[0]; a.b2 = blkN[b].b[1]; a.b3 = blkN[b].b[2];
      a.g = blkN[b].g; a.beta = blkN[b].beta;
      a.node = nodeL; a.agg = agg; a.out_bf16 = nodeL;
      a.nRows = NN;
      mlp_kernel<3, 256, 8, true><<<gN, 128, 0, stream>>>(a);
    }
  }

  // ---- decoder ----
  {
    MlpArgs a{};
    a.W1 = decw[0]; a.W2 = decw[1]; a.W3 = decw[2];
    a.b1 = dec.b[0]; a.b2 = dec.b[1]; a.b3 = decb3;
    a.node = nodeL; a.out_f32 = (float*)d_out; a.nRows = NN;
    mlp_kernel<4, 128, 1, false><<<gN, 128, 0, stream>>>(a);
  }
}
